// Stage_15934328668601
// MI455X (gfx1250) — matrix operand order repaired and checked by reading, not compiled, not run on hardware
//
#include <hip/hip_runtime.h>
#include <hip/hip_bf16.h>
#include <math.h>

// ---------------------------------------------------------------------------
// Types / helpers
// ---------------------------------------------------------------------------
typedef __attribute__((ext_vector_type(16))) __bf16 v16bf;
typedef __attribute__((ext_vector_type(8)))  float  v8f;

union Frag16 {
    uint4 q[2];
    v16bf v;
    unsigned short u[16];
};

__device__ __forceinline__ unsigned short f2bf(float f) {
    union { float f; unsigned int u; } x; x.f = f;
    unsigned int r = x.u + 0x7FFFu + ((x.u >> 16) & 1u);   // round-to-nearest-even
    return (unsigned short)(r >> 16);
}

__device__ __forceinline__ v8f wmma_bf16(const Frag16& a, const Frag16& b, v8f c) {
    return __builtin_amdgcn_wmma_f32_16x16x32_bf16(false, a.v, false, b.v,
                                                   (short)0, c, false, false);
}

// issue one 16B global->LDS async copy (GV mode), tracked by ASYNCcnt
__device__ __forceinline__ void async_b128(unsigned lds_off, const void* gaddr) {
    asm volatile("global_load_async_to_lds_b128 %0, %1, off"
                 :: "v"(lds_off), "v"((unsigned long long)(uintptr_t)gaddr)
                 : "memory");
}
__device__ __forceinline__ void wait_async0() {
    asm volatile("s_wait_asynccnt 0" ::: "memory");
}

// Problem constants
#define BB   16
#define NTOK 196
#define SPLN 98
#define AN   491          // 1 + 4*98 + 98
#define CC   768
#define HH   12
#define DD   64
#define HM   3072
#define MROWS (BB * AN)   // 7856
#define NP   512          // padded attention length
#define KSTR 72           // LDS K row stride (ushort)
#define VSTR 520          // LDS V^T row stride (ushort)

// ---------------------------------------------------------------------------
// Generic bf16 WMMA GEMM: out = act(A(MxK) @ W(KxN) + bias) [+ residual]
// Block tile 128x128, 8 waves (4x2), wave tile 32x64, K-tile 64 (2 wmma steps).
// A tile staged with GLOBAL_LOAD_ASYNC_TO_LDS_B128; W tile transposed in LDS.
// N must be a multiple of 128, K a multiple of 64. M may be ragged.
// ---------------------------------------------------------------------------
#define TM 128
#define TN 128
#define KT 64
#define ASTR (KT + 8)     // 72 ushorts = 144B rows, 16B aligned

__global__ __launch_bounds__(256)
void gemm_bf16(const unsigned short* __restrict__ A,
               const unsigned short* __restrict__ W,
               const float* __restrict__ bias,
               const float* __restrict__ residual,
               void* __restrict__ out, int out_is_bf16,
               int M, int N, int K, int act_gelu)
{
    __shared__ unsigned short As [TM][ASTR];
    __shared__ unsigned short Bst[TN][ASTR];   // W tile transposed: Bst[n][k]

    const int tid  = threadIdx.x;
    const int wave = tid >> 5;
    const int lane = tid & 31;
    const int wm   = wave >> 1;           // 0..3
    const int wn   = wave & 1;            // 0..1
    const int l16  = lane & 15;
    const int half = lane >> 4;
    const int bm   = blockIdx.x * TM;
    const int bn   = blockIdx.y * TN;

    v8f acc[2][4];
#pragma unroll
    for (int mi = 0; mi < 2; ++mi)
#pragma unroll
        for (int ni = 0; ni < 4; ++ni) acc[mi][ni] = v8f{};

    // per-thread staging coordinates
    const int arow = tid >> 1;            // 0..127
    const int acol = (tid & 1) * 32;      // 0 / 32
    int agr = bm + arow; if (agr >= M) agr = M - 1;   // clamp: junk rows discarded
    const unsigned lds_a = (unsigned)(uintptr_t)&As[arow][acol];
    const int wkk = tid >> 2;             // 0..63
    const int wnn = (tid & 3) * 32;       // 0..96

    for (int k0 = 0; k0 < K; k0 += KT) {
        // ---- A tile: 128 x 64 bf16 via async copy (4 x 16B per thread) ----
        {
            const unsigned short* gp = A + (size_t)agr * K + k0 + acol;
#pragma unroll
            for (int j = 0; j < 4; ++j)
                async_b128(lds_a + 16u * j, gp + 8 * j);
        }
        // ---- W tile: 64 x 128, register round-trip + transposed scatter ----
        {
            const uint4* src = (const uint4*)(W + (size_t)(k0 + wkk) * N + bn + wnn);
            unsigned short tmp[32];
            *(uint4*)&tmp[0]  = src[0];
            *(uint4*)&tmp[8]  = src[1];
            *(uint4*)&tmp[16] = src[2];
            *(uint4*)&tmp[24] = src[3];
#pragma unroll
            for (int i = 0; i < 32; ++i) Bst[wnn + i][wkk] = tmp[i];
        }
        // prefetch next W k-tile into cache while we compute
        if (k0 + KT < K)
            __builtin_prefetch(W + (size_t)(k0 + KT + wkk) * N + bn + wnn, 0, 1);

        wait_async0();
        __syncthreads();

        // ---- 2 k-steps x 8 WMMAs ----
#pragma unroll
        for (int ks = 0; ks < 2; ++ks) {
            Frag16 a[2], b[4];
            const int kbase = ks * 32 + half * 8;
#pragma unroll
            for (int mi = 0; mi < 2; ++mi) {
                int r = wm * 32 + mi * 16 + l16;
                a[mi].q[0] = *(const uint4*)&As[r][kbase];
                a[mi].q[1] = *(const uint4*)&As[r][kbase + 16];
            }
            const int koff = ks * 32 + half * 8;
#pragma unroll
            for (int ni = 0; ni < 4; ++ni) {
                int c = wn * 64 + ni * 16 + l16;
                b[ni].q[0] = *(const uint4*)&Bst[c][koff];
                b[ni].q[1] = *(const uint4*)&Bst[c][koff + 16];
            }
#pragma unroll
            for (int mi = 0; mi < 2; ++mi)
#pragma unroll
                for (int ni = 0; ni < 4; ++ni)
                    acc[mi][ni] = wmma_bf16(a[mi], b[ni], acc[mi][ni]);
        }
        __syncthreads();
    }

    // ---- epilogue ----
#pragma unroll
    for (int mi = 0; mi < 2; ++mi) {
#pragma unroll
        for (int ni = 0; ni < 4; ++ni) {
#pragma unroll
            for (int r = 0; r < 8; ++r) {
                int row = bm + wm * 32 + mi * 16 + half * 8 + r;
                int col = bn + wn * 64 + ni * 16 + l16;
                if (row < M) {
                    float v = acc[mi][ni][r] + bias[col];
                    if (act_gelu) v = 0.5f * v * (1.0f + erff(v * 0.70710678118f));
                    if (residual) v += residual[(size_t)row * N + col];
                    if (out_is_bf16)
                        ((unsigned short*)out)[(size_t)row * N + col] = f2bf(v);
                    else
                        ((float*)out)[(size_t)row * N + col] = v;
                }
            }
        }
    }
}

// ---------------------------------------------------------------------------
// Flash attention: one workgroup per (b,h). K (row-major) and V (transposed)
// for the whole head staged in LDS as bf16; streaming softmax in registers.
// ---------------------------------------------------------------------------
__global__ __launch_bounds__(256)
void flash_attn(const float* __restrict__ qkv, unsigned short* __restrict__ o)
{
    extern __shared__ unsigned short sm[];
    unsigned short* Kt = sm;                       // [512][KSTR]
    unsigned short* Vt = Kt + NP * KSTR;           // [64][VSTR] (V transposed)
    unsigned short* Ps = Vt + DD * VSTR;           // 8 waves * [16][40]

    const int bh = blockIdx.x;
    const int b  = bh / HH, h = bh % HH;
    const float* base = qkv + (size_t)b * AN * (3 * CC);
    const int tid = threadIdx.x;

    // cooperative K / V staging (zero-padded to 512 tokens)
    for (int idx = tid; idx < NP * DD; idx += 256) {
        int n = idx >> 6, d = idx & 63;
        float kv = 0.f, vv = 0.f;
        if (n < AN) {
            const float* rp = base + (size_t)n * (3 * CC) + h * DD + d;
            kv = rp[CC];        // K slab
            vv = rp[2 * CC];    // V slab
        }
        Kt[n * KSTR + d] = f2bf(kv);
        Vt[d * VSTR + n] = f2bf(vv);
    }
    __syncthreads();

    const int wave = tid >> 5, lane = tid & 31;
    const int l16 = lane & 15, half = lane >> 4;
    unsigned short* Pw = Ps + wave * (16 * 40);

    for (int rbi = 0; rbi < 2; ++rbi) {
        const int row0 = (wave + rbi * 8) * 32;    // 16 row-blocks of 32

        // Q fragments (loaded once per row block, from global f32)
        Frag16 qa[2][2];
#pragma unroll
        for (int mi = 0; mi < 2; ++mi) {
            int r = row0 + mi * 16 + l16;
            bool ok = r < AN;
            const float* qp = base + (size_t)(ok ? r : 0) * (3 * CC) + h * DD;
#pragma unroll
            for (int ks = 0; ks < 2; ++ks) {
                int kb = ks * 32 + half * 8;
#pragma unroll
                for (int j = 0; j < 8; ++j) {
                    qa[mi][ks].u[j]     = f2bf(ok ? qp[kb + j]      : 0.f);
                    qa[mi][ks].u[8 + j] = f2bf(ok ? qp[kb + 16 + j] : 0.f);
                }
            }
        }

        float mrow[2][8], lrow[2][8];
        v8f ov[2][4];
#pragma unroll
        for (int mi = 0; mi < 2; ++mi) {
#pragma unroll
            for (int r = 0; r < 8; ++r) { mrow[mi][r] = -3e38f; lrow[mi][r] = 0.f; }
#pragma unroll
            for (int dt = 0; dt < 4; ++dt) ov[mi][dt] = v8f{};
        }

        for (int j0 = 0; j0 < NP; j0 += 32) {
            // K^T fragments for 2 column tiles x 2 k-steps
            Frag16 kf[2][2];
#pragma unroll
            for (int nj = 0; nj < 2; ++nj) {
                int tok = j0 + nj * 16 + l16;
#pragma unroll
                for (int ks = 0; ks < 2; ++ks) {
                    int doff = ks * 32 + half * 8;
                    kf[nj][ks].q[0] = *(const uint4*)&Kt[tok * KSTR + doff];
                    kf[nj][ks].q[1] = *(const uint4*)&Kt[tok * KSTR + doff + 16];
                }
            }
#pragma unroll
            for (int mi = 0; mi < 2; ++mi) {
                v8f s0 = v8f{}, s1 = v8f{};
                s0 = wmma_bf16(qa[mi][0], kf[0][0], s0);
                s0 = wmma_bf16(qa[mi][1], kf[0][1], s0);
                s1 = wmma_bf16(qa[mi][0], kf[1][0], s1);
                s1 = wmma_bf16(qa[mi][1], kf[1][1], s1);

                const float msk0 = (j0 + l16      < AN) ? 0.f : -3e38f;
                const float msk1 = (j0 + 16 + l16 < AN) ? 0.f : -3e38f;
                float p0[8], p1[8];
#pragma unroll
                for (int r = 0; r < 8; ++r) {
                    p0[r] = s0[r] * 0.125f + msk0;
                    p1[r] = s1[r] * 0.125f + msk1;
                }
                // online softmax: row stats live across 16 lanes of a half-wave
#pragma unroll
                for (int r = 0; r < 8; ++r) {
                    float c = fmaxf(p0[r], p1[r]);
#pragma unroll
                    for (int off = 1; off < 16; off <<= 1)
                        c = fmaxf(c, __shfl_xor(c, off, 32));
                    float nm    = fmaxf(mrow[mi][r], c);
                    float alpha = __expf(mrow[mi][r] - nm);
                    mrow[mi][r] = nm;
                    p0[r] = __expf(p0[r] - nm);
                    p1[r] = __expf(p1[r] - nm);
                    float rs = p0[r] + p1[r];
#pragma unroll
                    for (int off = 1; off < 16; off <<= 1)
                        rs += __shfl_xor(rs, off, 32);
                    lrow[mi][r] = lrow[mi][r] * alpha + rs;
#pragma unroll
                    for (int dt = 0; dt < 4; ++dt) ov[mi][dt][r] *= alpha;
                }
                // P: C-layout -> A-layout via per-wave LDS pad
#pragma unroll
                for (int r = 0; r < 8; ++r) {
                    int pr = half * 8 + r;
                    Pw[pr * 40 + l16]      = f2bf(p0[r]);
                    Pw[pr * 40 + 16 + l16] = f2bf(p1[r]);
                }
                asm volatile("s_wait_dscnt 0" ::: "memory");
                Frag16 pa;
                pa.q[0] = *(const uint4*)&Pw[l16 * 40 + half * 8];
                pa.q[1] = *(const uint4*)&Pw[l16 * 40 + half * 8 + 16];
                // O += P(16x32) @ V(32x64)
#pragma unroll
                for (int dt = 0; dt < 4; ++dt) {
                    Frag16 vb;
                    int d = dt * 16 + l16;
                    int kof = j0 + half * 8;
                    vb.q[0] = *(const uint4*)&Vt[d * VSTR + kof];
                    vb.q[1] = *(const uint4*)&Vt[d * VSTR + kof + 16];
                    ov[mi][dt] = wmma_bf16(pa, vb, ov[mi][dt]);
                }
            }
        }
        // finalize: O /= l, store bf16 at o[b][row][h*64+d]
#pragma unroll
        for (int mi = 0; mi < 2; ++mi) {
#pragma unroll
            for (int r = 0; r < 8; ++r) {
                float inv = 1.0f / lrow[mi][r];
                int row = row0 + mi * 16 + half * 8 + r;
                if (row < AN) {
#pragma unroll
                    for (int dt = 0; dt < 4; ++dt)
                        o[((size_t)(b * AN + row)) * CC + h * DD + dt * 16 + l16] =
                            f2bf(ov[mi][dt][r] * inv);
                }
            }
        }
    }
}

// ---------------------------------------------------------------------------
// Small kernels
// ---------------------------------------------------------------------------
__global__ __launch_bounds__(256)
void layernorm_bf16(const float* __restrict__ x, const float* __restrict__ g,
                    const float* __restrict__ b, unsigned short* __restrict__ h)
{
    __shared__ float rs[256], rss[256];
    const int row = blockIdx.x;
    const float* xr = x + (size_t)row * CC;
    float s = 0.f, ss = 0.f;
    for (int i = threadIdx.x; i < CC; i += 256) { float v = xr[i]; s += v; ss += v * v; }
    rs[threadIdx.x] = s; rss[threadIdx.x] = ss; __syncthreads();
    for (int st = 128; st > 0; st >>= 1) {
        if ((int)threadIdx.x < st) { rs[threadIdx.x] += rs[threadIdx.x + st];
                                     rss[threadIdx.x] += rss[threadIdx.x + st]; }
        __syncthreads();
    }
    float mean = rs[0] * (1.0f / CC);
    float var  = rss[0] * (1.0f / CC) - mean * mean;
    float rstd = rsqrtf(var + 1e-6f);
    for (int i = threadIdx.x; i < CC; i += 256)
        h[(size_t)row * CC + i] = f2bf((xr[i] - mean) * rstd * g[i] + b[i]);
}

__global__ __launch_bounds__(256)
void sort_order(const float* __restrict__ attn, int* __restrict__ order)
{
    __shared__ float v[NTOK];
    const int b = blockIdx.x;
    if (threadIdx.x < NTOK) v[threadIdx.x] = attn[b * NTOK + threadIdx.x];
    __syncthreads();
    if (threadIdx.x < NTOK) {
        float mine = v[threadIdx.x];
        int r = 0;
        for (int j = 0; j < NTOK; ++j) {
            float o = v[j];
            r += (o > mine) || (o == mine && j < (int)threadIdx.x);
        }
        order[b * NTOK + r] = threadIdx.x;
    }
}

// cls row, sorted tail into x0, first 98 sorted tokens into bf16 GEMM-A
__global__ __launch_bounds__(256)
void build_inputs(const float* __restrict__ x, const int* __restrict__ order,
                  float* __restrict__ x0, unsigned short* __restrict__ Asp)
{
    const int g = blockIdx.x;           // B * 197 blocks
    const int b = g / 197, r = g % 197;
    if (r == 0) {
        const float* sp = x + (size_t)b * 197 * CC;
        float* dp = x0 + (size_t)b * AN * CC;
        for (int i = threadIdx.x; i < CC; i += 256) dp[i] = sp[i];
    } else {
        const int t = r - 1;
        const int src = order[b * NTOK + t];
        const float* sp = x + ((size_t)b * 197 + 1 + src) * CC;
        if (t < SPLN) {
            unsigned short* dp = Asp + ((size_t)b * SPLN + t) * CC;
            for (int i = threadIdx.x; i < CC; i += 256) dp[i] = f2bf(sp[i]);
        } else {
            float* dp = x0 + ((size_t)b * AN + 1 + 4 * SPLN + (t - SPLN)) * CC;
            for (int i = threadIdx.x; i < CC; i += 256) dp[i] = sp[i];
        }
    }
}

// st (B*98 x 3072) -> x0 rows 1..392 per batch (reshape is a contiguous copy)
__global__ void scatter_split(const float* __restrict__ st, float* __restrict__ x0)
{
    size_t i = (size_t)blockIdx.x * 256 + threadIdx.x;
    const size_t per = (size_t)SPLN * 4 * CC;            // 301056
    if (i < (size_t)BB * per) {
        size_t b = i / per, rem = i % per;
        x0[((size_t)b * AN + 1) * CC + rem] = st[i];
    }
}

__global__ void cvt_bf16(const float* __restrict__ s, unsigned short* __restrict__ d, long n)
{
    for (long i = (long)blockIdx.x * 256 + threadIdx.x; i < n; i += (long)gridDim.x * 256)
        d[i] = f2bf(s[i]);
}

__global__ void zero_f32(float* __restrict__ p, long n)
{
    for (long i = (long)blockIdx.x * 256 + threadIdx.x; i < n; i += (long)gridDim.x * 256)
        p[i] = 0.f;
}

// cls-row attention probabilities; accumulates mean-over-heads into gtmp
__global__ __launch_bounds__(256)
void cls_attn(const float* __restrict__ qkv, float* __restrict__ gtmp)
{
    __shared__ float q0[DD];
    __shared__ float sv[NP];
    __shared__ float red[256];
    const int bh = blockIdx.x, b = bh / HH, h = bh % HH;
    const float* base = qkv + (size_t)b * AN * (3 * CC);
    const int tid = threadIdx.x;
    if (tid < DD) q0[tid] = base[h * DD + tid];
    __syncthreads();
    for (int j = tid; j < NP; j += 256) {
        float s = -3e38f;
        if (j < AN) {
            const float* kr = base + (size_t)j * (3 * CC) + CC + h * DD;
            float acc = 0.f;
            for (int d = 0; d < DD; ++d) acc += q0[d] * kr[d];
            s = acc * 0.125f;
        }
        sv[j] = s;
    }
    __syncthreads();
    float m = -3e38f;
    for (int j = tid; j < NP; j += 256) m = fmaxf(m, sv[j]);
    red[tid] = m; __syncthreads();
    for (int st = 128; st > 0; st >>= 1) {
        if (tid < st) red[tid] = fmaxf(red[tid], red[tid + st]);
        __syncthreads();
    }
    const float mx = red[0]; __syncthreads();
    float s = 0.f;
    for (int j = tid; j < AN; j += 256) s += __expf(sv[j] - mx);
    red[tid] = s; __syncthreads();
    for (int st = 128; st > 0; st >>= 1) {
        if (tid < st) red[tid] += red[tid + st];
        __syncthreads();
    }
    const float inv = (1.0f / red[0]) * (1.0f / HH);
    for (int j = tid; j < AN; j += 256)
        if (j >= 1) atomicAdd(&gtmp[b * (AN - 1) + j - 1], __expf(sv[j] - mx) * inv);
}

__global__ void g_update(float* __restrict__ g, float* __restrict__ gtmp, long n)
{
    long i = (long)blockIdx.x * 256 + threadIdx.x;
    if (i < n) { g[i] = 0.5f * g[i] + 0.5f * gtmp[i]; gtmp[i] = 0.f; }
}

__global__ void final_copy(const float* __restrict__ x0, const float* __restrict__ g,
                           float* __restrict__ out)
{
    const long nx = (long)MROWS * CC;
    const long ng = (long)BB * (AN - 1);
    for (long i = (long)blockIdx.x * 256 + threadIdx.x; i < nx + ng;
         i += (long)gridDim.x * 256)
        out[i] = (i < nx) ? x0[i] : g[i - nx];
}

// ---------------------------------------------------------------------------
// Host orchestration
// ---------------------------------------------------------------------------
static inline dim3 gemm_grid(int M, int N) { return dim3((M + TM - 1) / TM, N / TN); }

extern "C" void kernel_launch(void* const* d_in, const int* in_sizes, int n_in,
                              void* d_out, int out_size, void* d_ws, size_t ws_size,
                              hipStream_t stream)
{
    (void)in_sizes; (void)n_in; (void)out_size; (void)ws_size;
    const float* x        = (const float*)d_in[0];
    const float* gattn    = (const float*)d_in[1];
    const float* split_w  = (const float*)d_in[2];
    const float* split_b  = (const float*)d_in[3];
    const float* ln1_g    = (const float*)d_in[4];
    const float* ln1_b    = (const float*)d_in[5];
    const float* qkv_w    = (const float*)d_in[6];
    const float* qkv_b    = (const float*)d_in[7];
    const float* proj_w   = (const float*)d_in[8];
    const float* proj_b   = (const float*)d_in[9];
    const float* ln2_g    = (const float*)d_in[10];
    const float* ln2_b    = (const float*)d_in[11];
    const float* fc1_w    = (const float*)d_in[12];
    const float* fc1_b    = (const float*)d_in[13];
    const float* fc2_w    = (const float*)d_in[14];
    const float* fc2_b    = (const float*)d_in[15];

    // workspace carve (256B aligned)
    char* p = (char*)d_ws;
    auto carve = [&](size_t bytes) -> void* {
        void* r = (void*)p; p += (bytes + 255) & ~(size_t)255; return r;
    };
    int*            order  = (int*)           carve((size_t)BB * NTOK * 4);
    unsigned short* Asp    = (unsigned short*)carve((size_t)BB * SPLN * CC * 2);
    float*          st     = (float*)         carve((size_t)BB * SPLN * 4 * CC * 4);
    float*          x0     = (float*)         carve((size_t)MROWS * CC * 4);
    unsigned short* hbf    = (unsigned short*)carve((size_t)MROWS * CC * 2);
    float*          qkv    = (float*)         carve((size_t)MROWS * 3 * CC * 4);
    unsigned short* obf    = (unsigned short*)carve((size_t)MROWS * CC * 2);
    unsigned short* ffn    = (unsigned short*)carve((size_t)MROWS * HM * 2);
    float*          gbuf   = (float*)         carve((size_t)BB * (AN - 1) * 4);
    float*          gtmp   = (float*)         carve((size_t)BB * (AN - 1) * 4);
    unsigned short* w_spl  = (unsigned short*)carve((size_t)CC * 4 * CC * 2);
    unsigned short* w_qkv  = (unsigned short*)carve((size_t)4 * CC * 3 * CC * 2);
    unsigned short* w_proj = (unsigned short*)carve((size_t)4 * CC * CC * 2);
    unsigned short* w_fc1  = (unsigned short*)carve((size_t)4 * CC * HM * 2);
    unsigned short* w_fc2  = (unsigned short*)carve((size_t)4 * HM * CC * 2);

    const int flash_lds = (NP * KSTR + DD * VSTR + 8 * 16 * 40) * 2;  // 150528 B
    hipFuncSetAttribute((const void*)flash_attn,
                        hipFuncAttributeMaxDynamicSharedMemorySize, flash_lds);

    // weight precision conversion (bf16 matrix-pipe path)
    cvt_bf16<<<1024, 256, 0, stream>>>(split_w, w_spl,  (long)CC * 4 * CC);
    cvt_bf16<<<1024, 256, 0, stream>>>(qkv_w,   w_qkv,  (long)4 * CC * 3 * CC);
    cvt_bf16<<<1024, 256, 0, stream>>>(proj_w,  w_proj, (long)4 * CC * CC);
    cvt_bf16<<<1024, 256, 0, stream>>>(fc1_w,   w_fc1,  (long)4 * CC * HM);
    cvt_bf16<<<1024, 256, 0, stream>>>(fc2_w,   w_fc2,  (long)4 * HM * CC);

    zero_f32<<<64, 256, 0, stream>>>(gbuf, (long)BB * (AN - 1));
    zero_f32<<<64, 256, 0, stream>>>(gtmp, (long)BB * (AN - 1));

    // token reorder + split projection + sequence assembly
    sort_order<<<BB, 256, 0, stream>>>(gattn, order);
    build_inputs<<<BB * 197, 256, 0, stream>>>(x, order, x0, Asp);
    gemm_bf16<<<gemm_grid(BB * SPLN, 4 * CC), 256, 0, stream>>>(
        Asp, w_spl, split_b, nullptr, st, 0, BB * SPLN, 4 * CC, CC, 0);
    scatter_split<<<(BB * SPLN * 4 * CC + 255) / 256, 256, 0, stream>>>(st, x0);

    for (int l = 0; l < 4; ++l) {
        layernorm_bf16<<<MROWS, 256, 0, stream>>>(x0, ln1_g + l * CC, ln1_b + l * CC, hbf);
        gemm_bf16<<<gemm_grid(MROWS, 3 * CC), 256, 0, stream>>>(
            hbf, w_qkv + (size_t)l * CC * 3 * CC, qkv_b + l * 3 * CC,
            nullptr, qkv, 0, MROWS, 3 * CC, CC, 0);
        flash_attn<<<BB * HH, 256, flash_lds, stream>>>(qkv, obf);
        cls_attn<<<BB * HH, 256, 0, stream>>>(qkv, gtmp);
        g_update<<<(BB * (AN - 1) + 255) / 256, 256, 0, stream>>>(
            gbuf, gtmp, (long)BB * (AN - 1));
        gemm_bf16<<<gemm_grid(MROWS, CC), 256, 0, stream>>>(
            obf, w_proj + (size_t)l * CC * CC, proj_b + l * CC,
            x0, x0, 0, MROWS, CC, CC, 0);
        layernorm_bf16<<<MROWS, 256, 0, stream>>>(x0, ln2_g + l * CC, ln2_b + l * CC, hbf);
        gemm_bf16<<<gemm_grid(MROWS, HM), 256, 0, stream>>>(
            hbf, w_fc1 + (size_t)l * CC * HM, fc1_b + l * HM,
            nullptr, ffn, 1, MROWS, HM, CC, 1);
        gemm_bf16<<<gemm_grid(MROWS, CC), 256, 0, stream>>>(
            ffn, w_fc2 + (size_t)l * HM * CC, fc2_b + l * CC,
            x0, x0, 0, MROWS, CC, HM, 0);
    }

    final_copy<<<2048, 256, 0, stream>>>(x0, gbuf, (float*)d_out);
}
